// ContrastiveLoss_10222022165007
// MI455X (gfx1250) — compile-verified, matrix operand order checked by reading, not run on hardware
//
#include <hip/hip_runtime.h>
#include <hip/hip_bf16.h>

// ---------------------------------------------------------------------------
// ContrastiveLoss on MI455X (gfx1250, wave32).
//
//   sim = l2norm(col) @ l2norm(row)^T   (8192x8192, K=1024)  -> 137 GFLOP
//   masked row reductions -> scalar
//
// Compute-bound => v_wmma_f32_16x16x32_f16 (f16 in, f32 accum).
// Normalized f16 operands (32 MB) live in d_ws, fully L2-resident.
// Wave computes a 32x32 tile (2x2 WMMA blocking). The K loop is manually
// software-pipelined (2 named buffer sets, prefetch distance 1) and kept
// NOT fully unrolled, so A-fragment LDS loads stay induction-based and the
// compiler cannot hoist/spill them to scratch (the round-2 failure mode).
// ---------------------------------------------------------------------------

#define NROW 8192
#define DDIM 1024
#define LDA  1032   // LDS row stride (halves): 1024 + 8 keeps 16B align, 4-bank/row rotation

typedef __attribute__((ext_vector_type(16))) _Float16 v16h;
typedef __attribute__((ext_vector_type(8)))  _Float16 v8h;
typedef __attribute__((ext_vector_type(4)))  _Float16 v4h;
typedef __attribute__((ext_vector_type(8)))  float    v8f;

union frag16 { v16h v; struct { v8h lo, hi; } s; };

#define WMMA_F16(Af, Bf, Cacc)                                               \
  __builtin_amdgcn_wmma_f32_16x16x32_f16(false, (Af).v, false, (Bf).v,       \
                                         (short)0, (Cacc), false, false)

// ---------------------------------------------------------------------------
// Kernel 1: L2-normalize one f32 row -> f16 row.  One block per row.
// ---------------------------------------------------------------------------
__global__ __launch_bounds__(256) void normalize_f16(const float* __restrict__ src,
                                                     _Float16* __restrict__ dst) {
  __shared__ float sred[8];
  const int row = blockIdx.x;
  const int tid = threadIdx.x;

  const float4 v = ((const float4*)(src + (size_t)row * DDIM))[tid]; // 256*4 = 1024
  float ss = v.x * v.x + v.y * v.y + v.z * v.z + v.w * v.w;

  #pragma unroll
  for (int m = 16; m >= 1; m >>= 1) ss += __shfl_xor(ss, m, 32);
  if ((tid & 31) == 0) sred[tid >> 5] = ss;
  __syncthreads();

  float tot = 0.0f;
  #pragma unroll
  for (int i = 0; i < 8; ++i) tot += sred[i];

  const float rn = 1.0f / sqrtf(tot + 1e-12f);   // matches x / sqrt(sum+1e-12)

  v4h out;
  out[0] = (_Float16)(v.x * rn);
  out[1] = (_Float16)(v.y * rn);
  out[2] = (_Float16)(v.z * rn);
  out[3] = (_Float16)(v.w * rn);
  ((v4h*)(dst + (size_t)row * DDIM))[tid] = out;   // 8-byte store
}

// ---------------------------------------------------------------------------
// Kernel 2: fused GEMM + masking + per-row reduction.
// Block (8 waves) owns rows [i0, i0+32); wave computes 32x32 tiles sweeping j.
// ---------------------------------------------------------------------------
__global__ __launch_bounds__(256) void sim_loss(const _Float16* __restrict__ A,  // colN f16 [N][D]
                                                const _Float16* __restrict__ B,  // rowN f16 [N][D]
                                                const int* __restrict__ tcol,
                                                const int* __restrict__ trow,
                                                float* __restrict__ row_final) {
  __shared__ _Float16 sA[32 * LDA];   // 66 KB A band (gfx1250: 320 KB/WGP)
  __shared__ float sLoss[32];
  __shared__ float sPos[32];

  const int tid  = threadIdx.x;
  const int wave = tid >> 5;
  const int lane = tid & 31;
  const int half = lane >> 4;   // selects K-chunk set per ISA 16-bit A/B striping
  const int l15  = lane & 15;
  const int hoff = half * 8;
  const int i0   = blockIdx.x * 32;

  if (tid < 32) { sLoss[tid] = 0.0f; sPos[tid] = 0.0f; }

  // Stage A band: 32 rows x 1024 halves, b128 chunks.
  for (int idx = tid; idx < 32 * (DDIM / 8); idx += 256) {
    const int r  = idx >> 7;        // idx / 128
    const int c8 = idx & 127;       // idx % 128
    *(v8h*)&sA[r * LDA + c8 * 8] =
        *(const v8h*)&A[(size_t)(i0 + r) * DDIM + c8 * 8];
  }
  __syncthreads();

  // targets_col for the 16 C-rows this lane produces (M = mi*16 + 8*half + r)
  int tci[2][8];
  #pragma unroll
  for (int mi = 0; mi < 2; ++mi)
    #pragma unroll
    for (int r = 0; r < 8; ++r)
      tci[mi][r] = tcol[i0 + mi * 16 + half * 8 + r];

  float accLoss[2][8], accPos[2][8];
  #pragma unroll
  for (int mi = 0; mi < 2; ++mi)
    #pragma unroll
    for (int r = 0; r < 8; ++r) { accLoss[mi][r] = 0.0f; accPos[mi][r] = 0.0f; }

  const _Float16* __restrict__ arow0 = &sA[l15 * LDA];
  const _Float16* __restrict__ arow1 = &sA[(16 + l15) * LDA];

  for (int jt = wave; jt < NROW / 32; jt += 8) {
    const int j0   = jt * 32;
    const int trj0 = trow[j0 + l15];
    const int trj1 = trow[j0 + 16 + l15];
    const _Float16* __restrict__ brow0 = B + (size_t)(j0 + l15) * DDIM;
    const _Float16* __restrict__ brow1 = B + (size_t)(j0 + 16 + l15) * DDIM;

    v8f c00 = (v8f){}, c01 = (v8f){}, c10 = (v8f){}, c11 = (v8f){};

    // Two named buffer sets (X, Y) -> pure SSA values, no arrays, no spills.
    frag16 a0X, a1X, b0X, b1X;
    frag16 a0Y, a1Y, b0Y, b1Y;

#define LOADF(K, A0, A1, B0, B1)                                             \
    do {                                                                     \
      (A0).s.lo = *(const v8h*)&arow0[(K) + hoff];        /* ds b128   */    \
      (A0).s.hi = *(const v8h*)&arow0[(K) + 16 + hoff];                      \
      (A1).s.lo = *(const v8h*)&arow1[(K) + hoff];                           \
      (A1).s.hi = *(const v8h*)&arow1[(K) + 16 + hoff];                      \
      (B0).s.lo = *(const v8h*)&brow0[(K) + hoff];        /* global b128 */  \
      (B0).s.hi = *(const v8h*)&brow0[(K) + 16 + hoff];                      \
      (B1).s.lo = *(const v8h*)&brow1[(K) + hoff];                           \
      (B1).s.hi = *(const v8h*)&brow1[(K) + 16 + hoff];                      \
    } while (0)

#define WMMA4(A0, A1, B0, B1)                                                \
    do {                                                                     \
      c00 = WMMA_F16((A0), (B0), c00);                                       \
      c01 = WMMA_F16((A0), (B1), c01);                                       \
      c10 = WMMA_F16((A1), (B0), c10);                                       \
      c11 = WMMA_F16((A1), (B1), c11);                                       \
    } while (0)

    LOADF(0, a0X, a1X, b0X, b1X);                 // prologue: group k=0 -> X
    #pragma unroll 1                              // keep induction-based addrs
    for (int k = 0; k < DDIM - 64; k += 64) {     // 15 iters: k = 0..896
      LOADF(k + 32, a0Y, a1Y, b0Y, b1Y);          // prefetch k+32 -> Y
      WMMA4(a0X, a1X, b0X, b1X);                  // consume k    (X)
      LOADF(k + 64, a0X, a1X, b0X, b1X);          // prefetch k+64 -> X
      WMMA4(a0Y, a1Y, b0Y, b1Y);                  // consume k+32 (Y)
    }
    // Tail: X holds group 960; load 992 -> Y; drain both.
    LOADF(DDIM - 32, a0Y, a1Y, b0Y, b1Y);
    WMMA4(a0X, a1X, b0X, b1X);
    WMMA4(a0Y, a1Y, b0Y, b1Y);
#undef LOADF
#undef WMMA4

    // Masked epilogue: element (M = mi*16 + r + 8*half, N = ni*16 + l15).
    #pragma unroll
    for (int r = 0; r < 8; ++r) {
      {
        const float s0 = c00[r];
        const bool same = (tci[0][r] == trj0);
        const bool pos  = same && (s0 < 1.0f - 1e-5f);
        const bool neg  = (!same) && (s0 > 0.5f);
        accLoss[0][r] += pos ? (1.0f - s0) : (neg ? s0 : 0.0f);
        accPos[0][r]  += pos ? 1.0f : 0.0f;
      }
      {
        const float s1 = c01[r];
        const bool same = (tci[0][r] == trj1);
        const bool pos  = same && (s1 < 1.0f - 1e-5f);
        const bool neg  = (!same) && (s1 > 0.5f);
        accLoss[0][r] += pos ? (1.0f - s1) : (neg ? s1 : 0.0f);
        accPos[0][r]  += pos ? 1.0f : 0.0f;
      }
      {
        const float s2 = c10[r];
        const bool same = (tci[1][r] == trj0);
        const bool pos  = same && (s2 < 1.0f - 1e-5f);
        const bool neg  = (!same) && (s2 > 0.5f);
        accLoss[1][r] += pos ? (1.0f - s2) : (neg ? s2 : 0.0f);
        accPos[1][r]  += pos ? 1.0f : 0.0f;
      }
      {
        const float s3 = c11[r];
        const bool same = (tci[1][r] == trj1);
        const bool pos  = same && (s3 < 1.0f - 1e-5f);
        const bool neg  = (!same) && (s3 > 0.5f);
        accLoss[1][r] += pos ? (1.0f - s3) : (neg ? s3 : 0.0f);
        accPos[1][r]  += pos ? 1.0f : 0.0f;
      }
    }
  }

  // One cross-lane reduction per block: sum over N (16 lanes within a half).
  #pragma unroll
  for (int mi = 0; mi < 2; ++mi) {
    #pragma unroll
    for (int r = 0; r < 8; ++r) {
      float l = accLoss[mi][r];
      float p = accPos[mi][r];
      #pragma unroll
      for (int m = 8; m >= 1; m >>= 1) {   // masks 1,2,4,8 stay inside each half
        l += __shfl_xor(l, m, 32);
        p += __shfl_xor(p, m, 32);
      }
      if (l15 == 0) {                      // lanes 0 and 16 carry the half-sums
        atomicAdd(&sLoss[mi * 16 + half * 8 + r], l);   // ds_add_f32
        atomicAdd(&sPos [mi * 16 + half * 8 + r], p);
      }
    }
  }
  __syncthreads();

  if (tid < 32) {
    const bool has = sPos[tid] > 0.0f;   // rows without positives are skipped
    row_final[i0 + tid] = has ? sLoss[tid] : 0.0f;
  }
}

// ---------------------------------------------------------------------------
// Kernel 3: sum 8192 row losses -> d_out[0] / n.  Single block.
// ---------------------------------------------------------------------------
__global__ __launch_bounds__(256) void final_reduce(const float* __restrict__ rf,
                                                    float* __restrict__ out) {
  __shared__ float sred[8];
  const int tid = threadIdx.x;
  float s = 0.0f;
  for (int i = tid; i < NROW; i += 256) s += rf[i];
  #pragma unroll
  for (int m = 16; m >= 1; m >>= 1) s += __shfl_xor(s, m, 32);
  if ((tid & 31) == 0) sred[tid >> 5] = s;
  __syncthreads();
  if (tid == 0) {
    float tot = 0.0f;
    #pragma unroll
    for (int i = 0; i < 8; ++i) tot += sred[i];
    out[0] = tot * (1.0f / (float)NROW);
  }
}

// ---------------------------------------------------------------------------
extern "C" void kernel_launch(void* const* d_in, const int* in_sizes, int n_in,
                              void* d_out, int out_size, void* d_ws, size_t ws_size,
                              hipStream_t stream) {
  const float* inputs_col  = (const float*)d_in[0];
  const int*   targets_col = (const int*)d_in[1];
  const float* inputs_row  = (const float*)d_in[2];
  const int*   target_row  = (const int*)d_in[3];

  char* ws = (char*)d_ws;
  _Float16* Ah = (_Float16*)ws;                                    // 16 MB
  _Float16* Bh = (_Float16*)(ws + (size_t)NROW * DDIM * 2);        // 16 MB
  float* row_final = (float*)(ws + (size_t)NROW * DDIM * 4);       // 32 KB

  normalize_f16<<<NROW, 256, 0, stream>>>(inputs_col, Ah);
  normalize_f16<<<NROW, 256, 0, stream>>>(inputs_row, Bh);
  sim_loss<<<NROW / 32, 256, 0, stream>>>(Ah, Bh, targets_col, target_row, row_final);
  final_reduce<<<1, 256, 0, stream>>>(row_final, (float*)d_out);
}
